// StatefulGRU_44693429682391
// MI455X (gfx1250) — matrix operand order repaired and checked by reading, not compiled, not run on hardware
//
#include <hip/hip_runtime.h>

// GRU: B=128, T=512, I=H=1024.  bf16 WMMA (v_wmma_f32_16x16x32_bf16), f32 accum.
// Phase 1: one big input-projection GEMM (parallel).
// Phase 2: single persistent kernel for the 512-step recurrent scan with a
//          device-wide barrier per step; w_hh tiles live in LDS for all steps.
#define BB 128
#define TT 512
#define II 1024
#define HH 1024
#define G3 (3 * HH)          // 3072
#define BT (BB * TT)         // 65536
#define NBLK 64              // blocks in the scan kernel (one per 16-col tile)
#define WPAD 1032            // padded LDS row stride in bf16 (kills bank conflicts)

typedef __attribute__((ext_vector_type(16))) __bf16 v16bf;
typedef __attribute__((ext_vector_type(8)))  __bf16 v8bf;
typedef __attribute__((ext_vector_type(8)))  float  v8f;

// ---------- helpers ----------
__device__ __forceinline__ __bf16 f32_to_bf16(float f) {
  union { float f; unsigned u; } a; a.f = f;
  unsigned r = a.u + 0x7FFFu + ((a.u >> 16) & 1u);   // round-to-nearest-even
  union { unsigned short s; __bf16 b; } c; c.s = (unsigned short)(r >> 16);
  return c.b;
}

__device__ __forceinline__ v16bf cat8(v8bf lo, v8bf hi) {
  return __builtin_shufflevector(lo, hi, 0,1,2,3,4,5,6,7,8,9,10,11,12,13,14,15);
}

// A fragment, 16x32 bf16, source row-major [M,K] with leading dim ld.
// lane L: row = L%16; VGPR v elem e: K = 16*(L/16) + 2v + e.
__device__ __forceinline__ v16bf load_fragA(const __bf16* __restrict__ base,
                                            int row0, int ld, int k0) {
  int lane = threadIdx.x & 31;
  const __bf16* p = base + (size_t)(row0 + (lane & 15)) * ld + k0 + ((lane >> 4) << 4);
  v8bf lo = *(const v8bf*)p;          // K = off .. off+7   -> VGPR 0..3
  v8bf hi = *(const v8bf*)(p + 8);    // K = off+8..off+15  -> VGPR 4..7
  return cat8(lo, hi);
}

// B fragment, 32x16 bf16, sourced from B^T stored row-major [N,K] (= weight
// matrix [out,in]).  lane L: col = L%16; VGPR v elem e: K = 16*(L/16)+2v+e.
__device__ __forceinline__ v16bf load_fragB(const __bf16* __restrict__ base,
                                            int row0, int ld, int k0) {
  int lane = threadIdx.x & 31;
  const __bf16* p = base + (size_t)(row0 + (lane & 15)) * ld + k0 + ((lane >> 4) << 4);
  v8bf lo = *(const v8bf*)p;
  v8bf hi = *(const v8bf*)(p + 8);
  return cat8(lo, hi);
}

// Same B-fragment gather but from an LDS tile with padded row stride WPAD.
__device__ __forceinline__ v16bf load_fragB_lds(const __bf16* rowbase, int k0) {
  int lane = threadIdx.x & 31;
  const __bf16* p = rowbase + (lane & 15) * WPAD + k0 + ((lane >> 4) << 4);
  v8bf lo = *(const v8bf*)p;
  v8bf hi = *(const v8bf*)(p + 8);
  return cat8(lo, hi);
}

__device__ __forceinline__ v8f wmma_bf16(v16bf a, v16bf b, v8f c) {
  return __builtin_amdgcn_wmma_f32_16x16x32_bf16(false, a, false, b,
                                                 (short)0, c, false, false);
}

__device__ __forceinline__ float sigmoid_f(float x) {
  return 1.0f / (1.0f + __expf(-x));
}

// ---------- kernels ----------
__global__ void __launch_bounds__(256) cvt_f32_bf16_kernel(
    const float* __restrict__ in, __bf16* __restrict__ out, long n) {
  long i = (long)blockIdx.x * blockDim.x + threadIdx.x;
  if (i < n) out[i] = f32_to_bf16(in[i]);
}

__global__ void __launch_bounds__(256) init_state_kernel(
    float* __restrict__ hf0, float* __restrict__ hf1,
    __bf16* __restrict__ hb0, __bf16* __restrict__ hb1,
    unsigned* __restrict__ sync) {
  int i = blockIdx.x * blockDim.x + threadIdx.x;   // BB*HH elements
  __bf16 z = f32_to_bf16(0.0f);
  hf0[i] = 0.0f; hf1[i] = 0.0f; hb0[i] = z; hb1[i] = z;
  if (i == 0) *sync = 0u;                          // re-arm the grid barrier
}

// gi = X(bf16)[BT,II] * Wih(bf16)[G3,II]^T + b_ih  -> GI f32 [BT,G3]
// one wave computes a 16x64 output block (A fragment reused x4).
__global__ void __launch_bounds__(256) gemm_gi_kernel(
    const __bf16* __restrict__ X, const __bf16* __restrict__ W,
    const float* __restrict__ bih, float* __restrict__ GI) {
  int gw   = blockIdx.x * 8 + (threadIdx.x >> 5);  // global wave id
  int tm   = gw / (G3 / 64);                       // 0..4095  (M tile)
  int tg   = gw % (G3 / 64);                       // 0..47    (group of 4 N tiles)
  int lane = threadIdx.x & 31;
  int rowA = tm * 16;
  int rowB = tg * 64;

  v8f acc[4] = {v8f{}, v8f{}, v8f{}, v8f{}};
  for (int k0 = 0; k0 < II; k0 += 32) {
    __builtin_prefetch(X + (size_t)rowA * II + k0 + 128, 0, 1);
    v16bf a = load_fragA(X, rowA, II, k0);
#pragma unroll
    for (int j = 0; j < 4; ++j) {
      v16bf b = load_fragB(W, rowB + j * 16, II, k0);
      acc[j] = wmma_bf16(a, b, acc[j]);
    }
  }
  int ncol = lane & 15, mh = (lane >> 4) * 8;
#pragma unroll
  for (int j = 0; j < 4; ++j) {
    int col = rowB + j * 16 + ncol;
    float bias = bih[col];
#pragma unroll
    for (int e = 0; e < 8; ++e) {
      int row = rowA + e + mh;
      // written once, consumed once much later -> keep out of near caches
      __builtin_nontemporal_store(acc[j][e] + bias, &GI[(size_t)row * G3 + col]);
    }
  }
}

// Persistent recurrent scan: 64 blocks (block = 16-col tile jt, wave = batch
// tile tm).  w_hh rows for this jt (3 gates x 16 rows x 1024 K) are staged in
// LDS once and reused for all 512 steps.  Device-wide barrier between steps.
__global__ void __launch_bounds__(256) gru_scan_kernel(
    const __bf16* __restrict__ Whh, const float* __restrict__ bhh,
    const float* __restrict__ GI,   float* __restrict__ Y,
    float* __restrict__ HF0, float* __restrict__ HF1,
    __bf16* __restrict__ HB0, __bf16* __restrict__ HB1,
    float* __restrict__ Hfinal, unsigned* __restrict__ sync) {
  __shared__ __bf16 shW[3 * 16 * WPAD];            // ~99 KB of the 320 KB WGP LDS

  int jt   = blockIdx.x;                           // 0..63  hidden-col tile
  int wave = threadIdx.x >> 5;                     // 0..7   batch tile
  int lane = threadIdx.x & 31;

  // cooperative load of the 48 weight rows into LDS (loop-invariant over t)
  for (int idx = threadIdx.x; idx < 48 * (II / 8); idx += 256) {
    int row = idx >> 7;                            // 0..47
    int c8  = (idx & 127) * 8;                     // bf16 column chunk
    int g = row >> 4, rr = row & 15;
    const __bf16* src = Whh + (size_t)(g * HH + jt * 16 + rr) * HH + c8;
    *(v8bf*)&shW[(g * 16 + rr) * WPAD + c8] = *(const v8bf*)src;
  }
  __syncthreads();

  const __bf16* shR = &shW[0];
  const __bf16* shZ = &shW[16 * WPAD];
  const __bf16* shN = &shW[32 * WPAD];

  int rowA = wave * 16;
  int ncol = lane & 15, mh = (lane >> 4) * 8;
  int col  = jt * 16 + ncol;
  float br = bhh[col], bz = bhh[HH + col], bn = bhh[2 * HH + col];

  for (int t = 0; t < TT; ++t) {
    const __bf16* Hb  = (t & 1) ? HB1 : HB0;
    const float*  Hf  = (t & 1) ? HF1 : HF0;
    float*        HfO = (t & 1) ? HF0 : HF1;
    __bf16*       HbO = (t & 1) ? HB0 : HB1;

    v8f accR = {}, accZ = {}, accN = {};
    for (int k0 = 0; k0 < HH; k0 += 32) {
      v16bf a = load_fragA(Hb, rowA, HH, k0);
      accR = wmma_bf16(a, load_fragB_lds(shR, k0), accR);
      accZ = wmma_bf16(a, load_fragB_lds(shZ, k0), accZ);
      accN = wmma_bf16(a, load_fragB_lds(shN, k0), accN);
    }

#pragma unroll
    for (int e = 0; e < 8; ++e) {
      int b = rowA + e + mh;                        // batch index
      size_t gi = ((size_t)b * TT + t) * (size_t)G3;
      float i_r = __builtin_nontemporal_load(&GI[gi + col]);
      float i_z = __builtin_nontemporal_load(&GI[gi + HH + col]);
      float i_n = __builtin_nontemporal_load(&GI[gi + 2 * HH + col]);
      float r = sigmoid_f(i_r + accR[e] + br);
      float z = sigmoid_f(i_z + accZ[e] + bz);
      float n = tanhf(i_n + r * (accN[e] + bn));
      float hp = Hf[b * HH + col];
      float hn = (1.0f - z) * n + z * hp;
      __builtin_nontemporal_store(hn, &Y[((size_t)b * TT + t) * HH + col]);
      HfO[b * HH + col] = hn;
      HbO[b * HH + col] = f32_to_bf16(hn);
      if (t == TT - 1) Hfinal[b * HH + col] = hn;
    }

    // ---- device-wide barrier: release writes, spin on arrival counter ----
    __threadfence();                                // make h_{t+1} visible
    __syncthreads();
    if (threadIdx.x == 0) {
      __hip_atomic_fetch_add(sync, 1u, __ATOMIC_RELEASE, __HIP_MEMORY_SCOPE_AGENT);
      unsigned target = (unsigned)NBLK * (unsigned)(t + 1);
      while (__hip_atomic_load(sync, __ATOMIC_ACQUIRE, __HIP_MEMORY_SCOPE_AGENT) < target)
        __builtin_amdgcn_s_sleep(1);
    }
    __syncthreads();
    __threadfence();                                // acquire h_{t+1} from peers
  }
}

// ---------- workspace layout (bytes) ----------
#define OFF_X16   ((size_t)0)                          // BT*II bf16  = 128 MiB
#define OFF_WIH   (OFF_X16 + (size_t)BT * II * 2)      // G3*II bf16  =   6 MiB
#define OFF_WHH   (OFF_WIH + (size_t)G3 * II * 2)      // G3*HH bf16  =   6 MiB
#define OFF_GI    (OFF_WHH + (size_t)G3 * HH * 2)      // BT*G3 f32   = 768 MiB
#define OFF_HF0   (OFF_GI  + (size_t)BT * G3 * 4)
#define OFF_HF1   (OFF_HF0 + (size_t)BB * HH * 4)
#define OFF_HB0   (OFF_HF1 + (size_t)BB * HH * 4)
#define OFF_HB1   (OFF_HB0 + (size_t)BB * HH * 2)
#define OFF_SYNC  (OFF_HB1 + (size_t)BB * HH * 2)

extern "C" void kernel_launch(void* const* d_in, const int* in_sizes, int n_in,
                              void* d_out, int out_size, void* d_ws, size_t ws_size,
                              hipStream_t stream) {
  const float* x    = (const float*)d_in[0];
  const float* w_ih = (const float*)d_in[1];
  const float* w_hh = (const float*)d_in[2];
  const float* b_ih = (const float*)d_in[3];
  const float* b_hh = (const float*)d_in[4];
  float* ys     = (float*)d_out;                       // [B,T,H]
  float* hfinal = ys + (size_t)BB * TT * HH;           // [B,H]

  char* ws = (char*)d_ws;
  __bf16*   X16   = (__bf16*)(ws + OFF_X16);
  __bf16*   Wih16 = (__bf16*)(ws + OFF_WIH);
  __bf16*   Whh16 = (__bf16*)(ws + OFF_WHH);
  float*    GI    = (float*)(ws + OFF_GI);
  float*    HF0   = (float*)(ws + OFF_HF0);
  float*    HF1   = (float*)(ws + OFF_HF1);
  __bf16*   HB0   = (__bf16*)(ws + OFF_HB0);
  __bf16*   HB1   = (__bf16*)(ws + OFF_HB1);
  unsigned* sync  = (unsigned*)(ws + OFF_SYNC);

  // 1) convert operands to bf16
  {
    long n = (long)BT * II;
    cvt_f32_bf16_kernel<<<(unsigned)((n + 255) / 256), 256, 0, stream>>>(x, X16, n);
    n = (long)G3 * II;
    cvt_f32_bf16_kernel<<<(unsigned)((n + 255) / 256), 256, 0, stream>>>(w_ih, Wih16, n);
    n = (long)G3 * HH;
    cvt_f32_bf16_kernel<<<(unsigned)((n + 255) / 256), 256, 0, stream>>>(w_hh, Whh16, n);
  }

  // 2) h0 = 0, barrier counter = 0
  init_state_kernel<<<(BB * HH) / 256, 256, 0, stream>>>(HF0, HF1, HB0, HB1, sync);

  // 3) input projection GEMM: waves = (BT/16)*(G3/64); 8 waves per block
  {
    unsigned blocks = (unsigned)(((size_t)(BT / 16) * (G3 / 64)) / 8);   // 24576
    gemm_gi_kernel<<<blocks, 256, 0, stream>>>(X16, Wih16, b_ih, GI);
  }

  // 4) recurrent scan: one persistent kernel, internal per-step grid barrier
  gru_scan_kernel<<<NBLK, 256, 0, stream>>>(Whh16, b_hh, GI, ys,
                                            HF0, HF1, HB0, HB1, hfinal, sync);
}
